// NTXent_4312147165445
// MI455X (gfx1250) — compile-verified, matrix operand order checked by reading, not run on hardware
//
#include <hip/hip_runtime.h>

typedef _Float16 half8 __attribute__((ext_vector_type(8)));
typedef _Float16 v16h  __attribute__((ext_vector_type(16)));
typedef float    v8f   __attribute__((ext_vector_type(8)));

#define N_ROWS 4096
#define DIM    256
#define TWO_N  8192
#define INV_T  2.0f   // 1 / TEMPERATURE, TEMPERATURE = 0.5

// ---------------------------------------------------------------------------
// Kernel 1: row-normalize z = [z_i; z_j] into f16 workspace; zero denom & out.
// One block (256 threads) per row; D == 256 so one element per thread.
// ---------------------------------------------------------------------------
__global__ __launch_bounds__(256) void ntxent_normalize(
    const float* __restrict__ z_i, const float* __restrict__ z_j,
    _Float16* __restrict__ znh, float* __restrict__ denom,
    float* __restrict__ out)
{
    __shared__ float red[256];
    const int row = blockIdx.x;
    const int t   = threadIdx.x;

    float v = (row < N_ROWS) ? z_i[row * DIM + t]
                             : z_j[(row - N_ROWS) * DIM + t];
    red[t] = v * v;
    __syncthreads();
    #pragma unroll
    for (int s = 128; s > 0; s >>= 1) {
        if (t < s) red[t] += red[t + s];
        __syncthreads();
    }
    float norm = fmaxf(sqrtf(red[0]), 1e-8f);
    znh[row * DIM + t] = (_Float16)(v / norm);

    if (t == 0) denom[row] = 0.0f;
    if (t == 0 && row == 0) out[0] = 0.0f;   // must re-zero each launch
}

// ---------------------------------------------------------------------------
// Kernel 2: fused sim = zn @ zn^T -> exp(sim/T) -> full row sums (diagonal
// INCLUDED here; it is subtracted analytically in the loss kernel, which
// keeps this hot loop free of per-element compares/selects).
//
// Wave tile: 32 rows x 128 cols = 16 accumulator fragments (2 row-groups x
// 8 col-fragments). Per K-step (K=32): 2 A-fragment loads + 8 B-fragment
// loads feed 16 v_wmma_f32_16x16x32_f16 -> 1.25 b128-loads per wmma.
//
// Workgroup = 8 waves = 2 row-halves x 4 col-quarters -> 64 x 512 tile.
// Grid = (8192/64, 8192/512) = (128, 16).
//
// Fragment layout (CDNA5 16-bit A 16x32): lane L<16 holds row M=L with
// K = k0+{0..7} and k0+16+{0..7}; lane L>=16 holds K = k0+8+{0..7} and
// k0+24+{0..7}. B (32x16) mirrors this with lane = column; since
// B = zn^T, a B fragment loads exactly like an A fragment of zn rows.
// C/D layout: VGPR r holds (row r, col lane) for lanes 0-15 and
// (row r+8, col lane-16) for lanes 16-31.
// ---------------------------------------------------------------------------
__device__ inline v16h load_frag_f16(const _Float16* __restrict__ rowPtr,
                                     int kk, int hi)
{
    const half8 lo = *(const half8*)(rowPtr + kk + hi * 8);
    const half8 hh = *(const half8*)(rowPtr + kk + 16 + hi * 8);
    v16h r;
    #pragma unroll
    for (int i = 0; i < 8; ++i) { r[i] = lo[i]; r[i + 8] = hh[i]; }
    return r;
}

__global__ __launch_bounds__(256) void ntxent_simsum(
    const _Float16* __restrict__ znh, float* __restrict__ denom)
{
    const int tid    = threadIdx.x;
    const int lane   = tid & 31;
    const int wave   = tid >> 5;
    const int laneLo = lane & 15;
    const int hi     = lane >> 4;

    const int rowBase = blockIdx.x * 64  + (wave >> 2) * 32;   // gridDim.x = 128
    const int colBase = blockIdx.y * 512 + (wave & 3) * 128;   // gridDim.y = 16

    const _Float16* aRow0 = znh + (size_t)(rowBase + laneLo) * DIM;
    const _Float16* aRow1 = znh + (size_t)(rowBase + 16 + laneLo) * DIM;
    const _Float16* bRow[8];
    #pragma unroll
    for (int f = 0; f < 8; ++f)
        bRow[f] = znh + (size_t)(colBase + f * 16 + laneLo) * DIM;

    v8f c[2][8];
    #pragma unroll
    for (int g = 0; g < 2; ++g)
        #pragma unroll
        for (int f = 0; f < 8; ++f)
            #pragma unroll
            for (int i = 0; i < 8; ++i) c[g][f][i] = 0.0f;

    #pragma unroll
    for (int kk = 0; kk < DIM; kk += 32) {
        const v16h a0 = load_frag_f16(aRow0, kk, hi);
        const v16h a1 = load_frag_f16(aRow1, kk, hi);
        #pragma unroll
        for (int f = 0; f < 8; ++f) {
            const v16h b = load_frag_f16(bRow[f], kk, hi);
            c[0][f] = __builtin_amdgcn_wmma_f32_16x16x32_f16(
                          false, a0, false, b, (short)0, c[0][f], false, false);
            c[1][f] = __builtin_amdgcn_wmma_f32_16x16x32_f16(
                          false, a1, false, b, (short)0, c[1][f], false, false);
        }
    }

    // exp(sim/T) and per-row partial sums (diagonal kept; removed later).
    float rowSum[2][8];
    #pragma unroll
    for (int g = 0; g < 2; ++g)
        #pragma unroll
        for (int r = 0; r < 8; ++r) rowSum[g][r] = 0.0f;

    #pragma unroll
    for (int f = 0; f < 8; ++f)
        #pragma unroll
        for (int g = 0; g < 2; ++g)
            #pragma unroll
            for (int r = 0; r < 8; ++r)
                rowSum[g][r] += __expf(c[g][f][r] * INV_T);

    // Reduce across the 16 columns held by each 16-lane half.
    #pragma unroll
    for (int g = 0; g < 2; ++g)
        #pragma unroll
        for (int r = 0; r < 8; ++r) {
            #pragma unroll
            for (int m = 1; m < 16; m <<= 1)
                rowSum[g][r] += __shfl_xor(rowSum[g][r], m, 16);
        }

    if (laneLo == 0) {
        #pragma unroll
        for (int g = 0; g < 2; ++g)
            #pragma unroll
            for (int r = 0; r < 8; ++r)
                atomicAdd(&denom[rowBase + g * 16 + r + (hi ? 8 : 0)],
                          rowSum[g][r]);
    }
}

// ---------------------------------------------------------------------------
// Kernel 3: positives + self-similarity + final loss. One wave per row.
// pos(row)  = dot(zn[row], zn[row +- N])   (f32 accumulation over f16 zn)
// self(row) = dot(zn[row], zn[row])        (same values the WMMA consumed)
// denominator = denom_full[row] - exp(self/T)   (removes diagonal term)
// loss = -(pos/T - log(denominator)); mean-accumulated into out.
// ---------------------------------------------------------------------------
__global__ __launch_bounds__(256) void ntxent_loss(
    const _Float16* __restrict__ znh, const float* __restrict__ denom,
    float* __restrict__ out)
{
    const int wave = threadIdx.x >> 5;
    const int lane = threadIdx.x & 31;
    const int row  = blockIdx.x * 8 + wave;
    const int partner = (row < N_ROWS) ? row + N_ROWS : row - N_ROWS;

    const _Float16* a = znh + (size_t)row * DIM + lane * 8;
    const _Float16* b = znh + (size_t)partner * DIM + lane * 8;
    float accP = 0.0f;   // positive-pair dot
    float accS = 0.0f;   // self dot
    #pragma unroll
    for (int i = 0; i < 8; ++i) {
        const float av = (float)a[i];
        const float bv = (float)b[i];
        accP += av * bv;
        accS += av * av;
    }
    #pragma unroll
    for (int m = 1; m < 32; m <<= 1) {
        accP += __shfl_xor(accP, m, 32);
        accS += __shfl_xor(accS, m, 32);
    }

    if (lane == 0) {
        const float denominator = denom[row] - __expf(accS * INV_T);
        const float loss = -(accP * INV_T - logf(denominator));
        atomicAdd(out, loss * (1.0f / (float)TWO_N));
    }
}

// ---------------------------------------------------------------------------
extern "C" void kernel_launch(void* const* d_in, const int* in_sizes, int n_in,
                              void* d_out, int out_size, void* d_ws, size_t ws_size,
                              hipStream_t stream)
{
    (void)in_sizes; (void)n_in; (void)out_size; (void)ws_size;
    const float* z_i = (const float*)d_in[0];
    const float* z_j = (const float*)d_in[1];
    float* out = (float*)d_out;

    _Float16* znh   = (_Float16*)d_ws;                                  // 4 MB
    float*    denom = (float*)((char*)d_ws +
                      (size_t)TWO_N * DIM * sizeof(_Float16));          // 32 KB

    ntxent_normalize<<<TWO_N, 256, 0, stream>>>(z_i, z_j, znh, denom, out);

    dim3 grid(TWO_N / 64, TWO_N / 512);   // 128 x 16 workgroups
    ntxent_simsum<<<grid, 256, 0, stream>>>(znh, denom);

    ntxent_loss<<<TWO_N / 8, 256, 0, stream>>>(znh, denom, out);
}
